// MCA_block_80264348828116
// MI455X (gfx1250) — matrix operand order repaired and checked by reading, not compiled, not run on hardware
//
#include <hip/hip_runtime.h>
#include <stdint.h>

// ---------------------------------------------------------------- types
typedef _Float16 h16;
typedef __attribute__((ext_vector_type(8)))  _Float16 v8h;
typedef __attribute__((ext_vector_type(16))) _Float16 v16h;
typedef __attribute__((ext_vector_type(8)))  float    v8f;
typedef __attribute__((ext_vector_type(4)))  int      v4i;

#define NTOK  343
#define NPAD  352            // 22 * 16
#define QT_N  22
#define EMB   192
#define NHEAD 6
#define HD    32
#define NN    (NTOK * NTOK)  // 117649
#define SCALE 0.17677669529663687f  // 32^-0.5

#if defined(__AMDGCN__) && __has_builtin(__builtin_amdgcn_global_load_async_to_lds_b128) && __has_builtin(__builtin_amdgcn_s_wait_asynccnt)
#define CDNA5_ASYNC 1
#else
#define CDNA5_ASYNC 0
#endif

__device__ __forceinline__ v16h cat16(v8h a, v8h b) {
  return __builtin_shufflevector(a, b, 0,1,2,3,4,5,6,7,8,9,10,11,12,13,14,15);
}
// B fragment: lane holds K {p..p+7},{p+16..p+23} for one column (p = kbase + 8*hi)
__device__ __forceinline__ v16h load_b16(const h16* p) {
  return cat16(*(const v8h*)p, *(const v8h*)(p + 16));
}
__device__ __forceinline__ void async_cp16(const h16* g, h16* l) {
#if CDNA5_ASYNC
  __builtin_amdgcn_global_load_async_to_lds_b128(
      (__attribute__((address_space(1))) v4i*)g,
      (__attribute__((address_space(3))) v4i*)l, 0, 0);
#else
  *(v8h*)l = *(const v8h*)g;
#endif
}
__device__ __forceinline__ void async_wait0() {
#if CDNA5_ASYNC
  __builtin_amdgcn_s_wait_asynccnt(0);
#endif
}

// ------------------------------------------------ kernel 0: weights->f16, bias expand
__global__ __launch_bounds__(256) void k_prep(
    const float* __restrict__ qw, const float* __restrict__ kvw,
    const float* __restrict__ pw, const float* __restrict__ btab,
    const int* __restrict__ ridx,
    h16* __restrict__ qw16, h16* __restrict__ kvw16,
    h16* __restrict__ pw16, float* __restrict__ biasx) {
  const int WQ = EMB * EMB, WKV = 2 * EMB * EMB, WP = EMB * EMB, NB = NHEAD * NN;
  for (int i = blockIdx.x * blockDim.x + threadIdx.x; i < WQ + WKV + WP + NB;
       i += gridDim.x * blockDim.x) {
    if (i < WQ)                 qw16[i] = (h16)qw[i];
    else if (i < WQ + WKV)      kvw16[i - WQ] = (h16)kvw[i - WQ];
    else if (i < WQ + WKV + WP) pw16[i - WQ - WKV] = (h16)pw[i - WQ - WKV];
    else {
      int j = i - WQ - WKV - WP;
      int hh = j / NN, rem = j - hh * NN;
      biasx[j] = btab[ridx[rem] * NHEAD + hh];
    }
  }
}

// A fragment from f32 row-major x: lane row m = lane%16; K runs {klo..klo+7},{klo+16..klo+23}
__device__ __forceinline__ void load_xfrags(const float* xbase, int b, int qt, int col,
                                            int klo, bool slot_ok, v16h ax[6]) {
  const int m = qt * 16 + col;
  const bool valid = slot_ok && (m < NTOK);
  const float* xr = xbase + ((size_t)b * NTOK + (valid ? m : 0)) * EMB;
#pragma unroll
  for (int kc = 0; kc < 6; ++kc) {
    v8h lo = {}, hi8 = {};
    if (valid) {
      float4 a0 = *(const float4*)(xr + kc * 32 + klo);
      float4 a1 = *(const float4*)(xr + kc * 32 + klo + 4);
      float4 a2 = *(const float4*)(xr + kc * 32 + klo + 16);
      float4 a3 = *(const float4*)(xr + kc * 32 + klo + 20);
      lo[0]=(h16)a0.x; lo[1]=(h16)a0.y; lo[2]=(h16)a0.z; lo[3]=(h16)a0.w;
      lo[4]=(h16)a1.x; lo[5]=(h16)a1.y; lo[6]=(h16)a1.z; lo[7]=(h16)a1.w;
      hi8[0]=(h16)a2.x; hi8[1]=(h16)a2.y; hi8[2]=(h16)a2.z; hi8[3]=(h16)a2.w;
      hi8[4]=(h16)a3.x; hi8[5]=(h16)a3.y; hi8[6]=(h16)a3.z; hi8[7]=(h16)a3.w;
    }
    ax[kc] = cat16(lo, hi8);
  }
}

__device__ __forceinline__ void store_rowmajor(h16* hbase, int qt, int hi, int d,
                                               const v8f& acc, float scalef) {
#pragma unroll
  for (int r = 0; r < 8; ++r) {
    const int mr = qt * 16 + r + 8 * hi;
    hbase[(size_t)mr * HD + d] = (h16)(acc[r] * scalef);
  }
}
__device__ __forceinline__ void store_vt(h16* vbase, int qt, int hi, int d, const v8f& acc) {
  v8h pk;
#pragma unroll
  for (int r = 0; r < 8; ++r) pk[r] = (h16)acc[r];
  *(v8h*)(vbase + (size_t)d * NPAD + qt * 16 + 8 * hi) = pk;
}

// ------------------------------------------------ kernel 1: QKV projections (WMMA)
// A-fragments (<=3 query tiles) are register-resident; the nt loop stays rolled so
// weight B-fragments are loaded exactly once per n-tile (no loop-invariant spills).
__global__ __launch_bounds__(128, 1) void k_qkv(
    const float* __restrict__ xin, const float* __restrict__ xc,
    const h16* __restrict__ qw16, const h16* __restrict__ kvw16,
    h16* __restrict__ Q16, h16* __restrict__ K16, h16* __restrict__ VT16) {
  const int b = blockIdx.x;
  const int lane = threadIdx.x & 31, wid = threadIdx.x >> 5;
  const int col = lane & 15, hi = lane >> 4, klo = hi * 8;

  const int qt0 = wid + 4 * blockIdx.y;  // 0..7
  const int qt1 = qt0 + 8;               // 8..15 (always valid)
  const int qt2 = qt0 + 16;              // 16..23 (valid iff < 22)
  const bool v2 = qt2 < QT_N;

  // ---------------- phase Q: x_in @ q_w^T (scaled)
  {
    v16h A0[6], A1[6], A2[6];
    load_xfrags(xin, b, qt0, col, klo, true, A0);
    load_xfrags(xin, b, qt1, col, klo, true, A1);
    load_xfrags(xin, b, qt2, col, klo, v2, A2);
#pragma unroll 1
    for (int nt = 0; nt < 12; ++nt) {
      const int n = nt * 16 + col;
      v16h BW[6];
#pragma unroll
      for (int kc = 0; kc < 6; ++kc)
        BW[kc] = load_b16(qw16 + (size_t)n * EMB + kc * 32 + hi * 8);
      const int hh = nt >> 1, d = (nt & 1) * 16 + col;
      h16* qb = Q16 + (size_t)(b * NHEAD + hh) * NPAD * HD;

      v8f a0 = {}, a1 = {}, a2 = {};
#pragma unroll
      for (int kc = 0; kc < 6; ++kc)
        a0 = __builtin_amdgcn_wmma_f32_16x16x32_f16(false, A0[kc], false, BW[kc], (short)0, a0, false, false);
#pragma unroll
      for (int kc = 0; kc < 6; ++kc)
        a1 = __builtin_amdgcn_wmma_f32_16x16x32_f16(false, A1[kc], false, BW[kc], (short)0, a1, false, false);
#pragma unroll
      for (int kc = 0; kc < 6; ++kc)
        a2 = __builtin_amdgcn_wmma_f32_16x16x32_f16(false, A2[kc], false, BW[kc], (short)0, a2, false, false);
      store_rowmajor(qb, qt0, hi, d, a0, SCALE);
      store_rowmajor(qb, qt1, hi, d, a1, SCALE);
      if (v2) store_rowmajor(qb, qt2, hi, d, a2, SCALE);
    }
  }
  // ---------------- phase KV: x_cross @ kv_w^T
  {
    v16h A0[6], A1[6], A2[6];
    load_xfrags(xc, b, qt0, col, klo, true, A0);
    load_xfrags(xc, b, qt1, col, klo, true, A1);
    load_xfrags(xc, b, qt2, col, klo, v2, A2);
#pragma unroll 1
    for (int nt = 0; nt < 24; ++nt) {
      const int n = nt * 16 + col;
      v16h BW[6];
#pragma unroll
      for (int kc = 0; kc < 6; ++kc)
        BW[kc] = load_b16(kvw16 + (size_t)n * EMB + kc * 32 + hi * 8);

      v8f a0 = {}, a1 = {}, a2 = {};
#pragma unroll
      for (int kc = 0; kc < 6; ++kc)
        a0 = __builtin_amdgcn_wmma_f32_16x16x32_f16(false, A0[kc], false, BW[kc], (short)0, a0, false, false);
#pragma unroll
      for (int kc = 0; kc < 6; ++kc)
        a1 = __builtin_amdgcn_wmma_f32_16x16x32_f16(false, A1[kc], false, BW[kc], (short)0, a1, false, false);
#pragma unroll
      for (int kc = 0; kc < 6; ++kc)
        a2 = __builtin_amdgcn_wmma_f32_16x16x32_f16(false, A2[kc], false, BW[kc], (short)0, a2, false, false);

      if (nt < 12) {  // K: row-major [h][n][32]
        const int hh = nt >> 1, d = (nt & 1) * 16 + col;
        h16* kb = K16 + (size_t)(b * NHEAD + hh) * NPAD * HD;
        store_rowmajor(kb, qt0, hi, d, a0, 1.0f);
        store_rowmajor(kb, qt1, hi, d, a1, 1.0f);
        if (v2) store_rowmajor(kb, qt2, hi, d, a2, 1.0f);
      } else {        // V: transposed [h][32][n] -> contiguous 16B stores
        const int vh = (nt - 12) >> 1, d = ((nt - 12) & 1) * 16 + col;
        h16* vb = VT16 + (size_t)(b * NHEAD + vh) * HD * NPAD;
        store_vt(vb, qt0, hi, d, a0);
        store_vt(vb, qt1, hi, d, a1);
        if (v2) store_vt(vb, qt2, hi, d, a2);
      }
    }
  }
}

// ------------------------------------------------ kernel 2: fused attention per (b,h)
__global__ __launch_bounds__(128, 1) void k_attn(
    const h16* __restrict__ Q16, const h16* __restrict__ K16,
    const h16* __restrict__ VT16, const float* __restrict__ biasx,
    const float* __restrict__ mask, h16* __restrict__ O16) {
  __shared__ __align__(16) h16 sK[NPAD][40];   // K rows padded: 80B stride (16B-mult, bank-clean)
  __shared__ __align__(16) h16 sV[HD][360];    // Vt rows padded: 720B stride
  __shared__ __align__(16) h16 sP[4][16][56];  // per-wave P transpose: 112B stride

  const int bx = blockIdx.x;
  const int b = bx / NHEAD, h = bx - b * NHEAD;
  const int tid = threadIdx.x, lane = tid & 31, wid = tid >> 5;
  const int col = lane & 15, hi = lane >> 4;

  const h16* Kg = K16 + (size_t)bx * NPAD * HD;
  const h16* Vg = VT16 + (size_t)bx * HD * NPAD;
  // stage K (352x32) and Vt (32x352) into padded LDS (async if available)
  for (int c = tid; c < NPAD * 4; c += 128) {
    int row = c >> 2, p = c & 3;
    async_cp16(Kg + (size_t)row * HD + p * 8, &sK[row][p * 8]);
  }
  for (int c = tid; c < HD * 44; c += 128) {
    int row = c / 44, p = c - row * 44;
    async_cp16(Vg + (size_t)row * NPAD + p * 8, &sV[row][p * 8]);
  }
  async_wait0();
  __syncthreads();

  const float* bxp = biasx + (size_t)h * NN;
  const float* mkp = mask + (size_t)(b & 63) * NN;
  const int klo = hi * 8;

  for (int qt = wid; qt < QT_N; qt += 4) {
    // Q A-fragment (registers, reused across all key tiles)
    const h16* Qr = Q16 + ((size_t)bx * NPAD + qt * 16 + col) * HD;
    v16h aq = cat16(*(const v8h*)(Qr + klo), *(const v8h*)(Qr + klo + 16));

    v8f acc[QT_N];
#pragma unroll
    for (int kt = 0; kt < QT_N; ++kt) {
      const h16* kp = &sK[kt * 16 + col][hi * 8];
      v16h bk = cat16(*(const v8h*)kp, *(const v8h*)(kp + 16));
      v8f z = {};
      acc[kt] = __builtin_amdgcn_wmma_f32_16x16x32_f16(false, aq, false, bk,
                                                       (short)0, z, false, false);
    }
    // bias + window mask + key-padding mask
#pragma unroll
    for (int kt = 0; kt < QT_N; ++kt) {
      const int k = kt * 16 + col;
#pragma unroll
      for (int r = 0; r < 8; ++r) {
        const int q = qt * 16 + r + 8 * hi;
        if (k < NTOK) {
          if (q < NTOK) acc[kt][r] += bxp[q * NTOK + k] + mkp[q * NTOK + k];
        } else {
          acc[kt][r] = -1e30f;
        }
      }
    }
    // row-wise softmax: per-lane partial, then 16-lane-half shuffle reduce
    float mx[8], sm[8];
#pragma unroll
    for (int r = 0; r < 8; ++r) {
      float v = -1e30f;
#pragma unroll
      for (int kt = 0; kt < QT_N; ++kt) v = fmaxf(v, acc[kt][r]);
      v = fmaxf(v, __shfl_xor(v, 1, 32)); v = fmaxf(v, __shfl_xor(v, 2, 32));
      v = fmaxf(v, __shfl_xor(v, 4, 32)); v = fmaxf(v, __shfl_xor(v, 8, 32));
      mx[r] = v;
    }
#pragma unroll
    for (int kt = 0; kt < QT_N; ++kt)
#pragma unroll
      for (int r = 0; r < 8; ++r) acc[kt][r] = __expf(acc[kt][r] - mx[r]);
#pragma unroll
    for (int r = 0; r < 8; ++r) {
      float s = 0.f;
#pragma unroll
      for (int kt = 0; kt < QT_N; ++kt) s += acc[kt][r];
      s += __shfl_xor(s, 1, 32); s += __shfl_xor(s, 2, 32);
      s += __shfl_xor(s, 4, 32); s += __shfl_xor(s, 8, 32);
      sm[r] = s;
    }
    // PV: transpose P (D-layout -> A-layout) through per-wave LDS, chained WMMA
    v8f vacc0 = {}, vacc1 = {};
#pragma unroll
    for (int kc = 0; kc < 11; ++kc) {
#pragma unroll
      for (int t = 0; t < 2; ++t) {
        const int kt2 = 2 * kc + t;
#pragma unroll
        for (int r = 0; r < 8; ++r)
          sP[wid][r + 8 * hi][t * 16 + col] = (h16)acc[kt2][r];
      }
      const h16* pr = &sP[wid][col][klo];
      v16h ap = cat16(*(const v8h*)pr, *(const v8h*)(pr + 16));
      const h16* v0 = &sV[col][kc * 32 + hi * 8];
      const h16* v1 = &sV[16 + col][kc * 32 + hi * 8];
      v16h bv0 = cat16(*(const v8h*)v0, *(const v8h*)(v0 + 16));
      v16h bv1 = cat16(*(const v8h*)v1, *(const v8h*)(v1 + 16));
      vacc0 = __builtin_amdgcn_wmma_f32_16x16x32_f16(false, ap, false, bv0,
                                                     (short)0, vacc0, false, false);
      vacc1 = __builtin_amdgcn_wmma_f32_16x16x32_f16(false, ap, false, bv1,
                                                     (short)0, vacc1, false, false);
    }
    // epilogue: fold 1/rowsum, store O[b][n][c]
#pragma unroll
    for (int r = 0; r < 8; ++r) {
      const int m = qt * 16 + r + 8 * hi;
      if (m < NTOK) {
        const float inv = 1.0f / sm[r];
        h16* orow = O16 + ((size_t)b * NPAD + m) * EMB + h * HD;
        orow[col] = (h16)(vacc0[r] * inv);
        orow[16 + col] = (h16)(vacc1[r] * inv);
      }
    }
  }
}

// ------------------------------------------------ kernel 3: output projection
__global__ __launch_bounds__(128, 1) void k_proj(
    const h16* __restrict__ O16, const h16* __restrict__ pw16,
    const float* __restrict__ pb, float* __restrict__ out) {
  const int b = blockIdx.x;
  const int lane = threadIdx.x & 31, wid = threadIdx.x >> 5;
  const int col = lane & 15, hi = lane >> 4, klo = hi * 8;

  const int qt0 = wid + 4 * blockIdx.y;
  const int qt1 = qt0 + 8;
  const int qt2 = qt0 + 16;
  const bool v2 = qt2 < QT_N;

  v16h A0[6], A1[6], A2[6];
#pragma unroll
  for (int kc = 0; kc < 6; ++kc) {
    const h16* r0 = O16 + ((size_t)b * NPAD + qt0 * 16 + col) * EMB + kc * 32 + klo;
    const h16* r1 = O16 + ((size_t)b * NPAD + qt1 * 16 + col) * EMB + kc * 32 + klo;
    const h16* r2 = O16 + ((size_t)b * NPAD + (v2 ? qt2 : qt0) * 16 + col) * EMB + kc * 32 + klo;
    A0[kc] = cat16(*(const v8h*)r0, *(const v8h*)(r0 + 16));
    A1[kc] = cat16(*(const v8h*)r1, *(const v8h*)(r1 + 16));
    A2[kc] = cat16(*(const v8h*)r2, *(const v8h*)(r2 + 16));
  }
#pragma unroll 1
  for (int nt = 0; nt < 12; ++nt) {
    const int n = nt * 16 + col;
    v16h BW[6];
#pragma unroll
    for (int kc = 0; kc < 6; ++kc)
      BW[kc] = load_b16(pw16 + (size_t)n * EMB + kc * 32 + hi * 8);

    v8f a0 = {}, a1 = {}, a2 = {};
#pragma unroll
    for (int kc = 0; kc < 6; ++kc)
      a0 = __builtin_amdgcn_wmma_f32_16x16x32_f16(false, A0[kc], false, BW[kc], (short)0, a0, false, false);
#pragma unroll
    for (int kc = 0; kc < 6; ++kc)
      a1 = __builtin_amdgcn_wmma_f32_16x16x32_f16(false, A1[kc], false, BW[kc], (short)0, a1, false, false);
#pragma unroll
    for (int kc = 0; kc < 6; ++kc)
      a2 = __builtin_amdgcn_wmma_f32_16x16x32_f16(false, A2[kc], false, BW[kc], (short)0, a2, false, false);

    const float bias = pb[n];
#pragma unroll
    for (int r = 0; r < 8; ++r) {
      const int m0 = qt0 * 16 + r + 8 * hi;
      const int m1 = qt1 * 16 + r + 8 * hi;
      if (m0 < NTOK) out[((size_t)b * NTOK + m0) * EMB + n] = a0[r] + bias;
      if (m1 < NTOK) out[((size_t)b * NTOK + m1) * EMB + n] = a1[r] + bias;
      if (v2) {
        const int m2 = qt2 * 16 + r + 8 * hi;
        if (m2 < NTOK) out[((size_t)b * NTOK + m2) * EMB + n] = a2[r] + bias;
      }
    }
  }
}

// ---------------------------------------------------------------- launcher
extern "C" void kernel_launch(void* const* d_in, const int* in_sizes, int n_in,
                              void* d_out, int out_size, void* d_ws, size_t ws_size,
                              hipStream_t stream) {
  const float* xin  = (const float*)d_in[0];
  const float* xc   = (const float*)d_in[1];
  const float* mask = (const float*)d_in[2];
  const float* qw   = (const float*)d_in[3];
  const float* kvw  = (const float*)d_in[4];
  const float* pw   = (const float*)d_in[5];
  const float* pb   = (const float*)d_in[6];
  const float* btab = (const float*)d_in[7];
  const int*   ridx = (const int*)d_in[8];

  char* w = (char*)d_ws;
  size_t off = 0;
  auto take = [&](size_t bytes) { size_t r = off; off = (off + bytes + 255) & ~(size_t)255; return r; };
  h16*   qw16  = (h16*)(w + take((size_t)EMB * EMB * 2));
  h16*   kvw16 = (h16*)(w + take((size_t)2 * EMB * EMB * 2));
  h16*   pw16  = (h16*)(w + take((size_t)EMB * EMB * 2));
  float* biasx = (float*)(w + take((size_t)NHEAD * NN * 4));
  h16*   Q16   = (h16*)(w + take((size_t)256 * NHEAD * NPAD * HD * 2));
  h16*   K16   = (h16*)(w + take((size_t)256 * NHEAD * NPAD * HD * 2));
  h16*   VT16  = (h16*)(w + take((size_t)256 * NHEAD * HD * NPAD * 2));
  h16*   O16   = (h16*)(w + take((size_t)256 * NPAD * EMB * 2));

  k_prep<<<3334, 256, 0, stream>>>(qw, kvw, pw, btab, ridx, qw16, kvw16, pw16, biasx);
  k_qkv<<<dim3(256, 2), 128, 0, stream>>>(xin, xc, qw16, kvw16, Q16, K16, VT16);
  k_attn<<<256 * NHEAD, 128, 0, stream>>>(Q16, K16, VT16, biasx, mask, O16);
  k_proj<<<dim3(256, 2), 128, 0, stream>>>(O16, pw16, pb, (float*)d_out);
}
